// DiagonnalyMaskedSelfAttention_66374424592827
// MI455X (gfx1250) — matrix operand order repaired and checked by reading, not compiled, not run on hardware
//
#include <hip/hip_runtime.h>
#include <hip/hip_fp16.h>

// ---- MI455X / gfx1250: wave32, WMMA f32_16x16x32_f16 ----
typedef __attribute__((ext_vector_type(16))) _Float16 v16h;
typedef __attribute__((ext_vector_type(8)))  _Float16 v8h;
typedef __attribute__((ext_vector_type(8)))  float    v8f;

#define WMMA16(a, b, c) \
  __builtin_amdgcn_wmma_f32_16x16x32_f16(false, (a), false, (b), (short)0, (c), false, false)

#define Bc   4
#define Lc   2048
#define DIMc 512
#define Hc   8
#define Dc   64
#define SCALEc 0.125f      /* 64^-0.5 */
#define NEGc  -10000.0f

// A-operand (16x32 f16, M x K) fragment from row-major [*, ld] matrix.
__device__ __forceinline__ v16h load_a_frag(const _Float16* __restrict__ A, int ld,
                                            long m0, int k0, int lane) {
  const int hi = lane >> 4;
  const _Float16* r = A + (m0 + (lane & 15)) * (long)ld;
  return *(const v16h*)(r + k0 + 16 * hi);
}

// B-operand (32x16 f16, K x N) for C = A * Wstore^T : one contiguous 32B load/lane.
__device__ __forceinline__ v16h load_bt_frag(const _Float16* __restrict__ W, int ld,
                                             long n0, int k0, int lane) {
  const int hi = lane >> 4;
  return *(const v16h*)(W + (n0 + (lane & 15)) * (long)ld + k0 + 16 * hi);
}

// ---------------- f32 -> f16 convert ----------------
__global__ void cvt_f32_f16(const float* __restrict__ in, _Float16* __restrict__ out, int n) {
  int i = blockIdx.x * 256 + threadIdx.x;
  if (i < n) out[i] = (_Float16)in[i];
}

// ---------------- WMMA GEMM:  C[M,N] = A[M,K] * W[N,K]^T ----------------
// MODE 0: f16 -> [B,H,L,64]   MODE 1: f16 -> [B,H,64,L]   MODE 2: f32 row-major
template <int MODE>
__global__ __launch_bounds__(128) void gemm_nt(const _Float16* __restrict__ A,
                                               const _Float16* __restrict__ W,
                                               void* __restrict__ outp,
                                               int M, int N, int K) {
  const int lane = threadIdx.x & 31;
  const int wave = threadIdx.x >> 5;
  const int tilesN = N >> 5;
  const int t  = blockIdx.x * 4 + wave;
  const long m0 = (long)(t / tilesN) * 32;
  const long n0 = (long)(t % tilesN) * 32;

  struct GF { v16h a0, a1, b0, b1; };
  v8f c00 = {}, c01 = {}, c10 = {}, c11 = {};

  auto gload = [&](GF& f, int k0) {
    f.a0 = load_a_frag(A, K, m0,      k0, lane);
    f.a1 = load_a_frag(A, K, m0 + 16, k0, lane);
    f.b0 = load_bt_frag(W, K, n0,      k0, lane);
    f.b1 = load_bt_frag(W, K, n0 + 16, k0, lane);
  };
  auto gstep = [&](int k0, GF& cur, GF& nxt) {
    const int kn = (k0 + 32 < K) ? (k0 + 32) : k0;
    gload(nxt, kn);
    c00 = WMMA16(cur.a0, cur.b0, c00);
    c01 = WMMA16(cur.a0, cur.b1, c01);
    c10 = WMMA16(cur.a1, cur.b0, c10);
    c11 = WMMA16(cur.a1, cur.b1, c11);
  };

  GF f0, f1;
  gload(f0, 0);
  for (int k0 = 0; k0 < K; k0 += 64) {   // ping-pong: no register rotation moves
    gstep(k0,      f0, f1);
    gstep(k0 + 32, f1, f0);
  }

  const int hi = lane >> 4, ln = lane & 15;
  v8f cc[2][2] = { {c00, c01}, {c10, c11} };
#pragma unroll
  for (int mi = 0; mi < 2; ++mi)
#pragma unroll
    for (int ni = 0; ni < 2; ++ni)
#pragma unroll
      for (int i = 0; i < 8; ++i) {
        const long r   = m0 + mi * 16 + i + 8 * hi;
        const long col = n0 + ni * 16 + ln;
        const float v  = cc[mi][ni][i];
        if (MODE == 2) {
          ((float*)outp)[r * N + col] = v;
        } else {
          const long b = r >> 11, l = r & 2047;
          const long h = col >> 6, d = col & 63;
          _Float16* o = (_Float16*)outp;
          if (MODE == 0)
            o[(((b * Hc + h) * Lc) + l) * Dc + d] = (_Float16)v;
          else
            o[(((b * Hc + h) * Dc) + d) * Lc + l] = (_Float16)v;
        }
      }
}

// ---------------- Fused flash attention with diagonal mask ----------------
// One wave handles 32 queries (two 16-q tiles sharing every K/V fragment).
// S^T = K*Q^T so the S^T C-layout equals P's A-operand layout lane-for-lane.
// K fragments: cross-iteration ping-pong double buffer (no rotation moves).
// V fragments: loaded at body top, consumed after S-WMMAs + softmax (latency hidden).
__global__ __launch_bounds__(128) void attn_diagmask(const _Float16* __restrict__ Q,
                                                     const _Float16* __restrict__ Km,
                                                     const _Float16* __restrict__ Vt,
                                                     _Float16* __restrict__ O) {
  const int lane = threadIdx.x & 31;
  const int wave = threadIdx.x >> 5;
  const int flat = blockIdx.x * 4 + wave;   // 0 .. B*H*(L/32)-1 = 2047
  const int bh = flat >> 6;                 // / (L/32)=64
  const int q0 = (flat & 63) * 32;
  const int hi = lane >> 4, ln = lane & 15;

  const _Float16* Qh = Q  + (long)bh * Lc * Dc;
  const _Float16* Kh = Km + (long)bh * Lc * Dc;
  const _Float16* Vh = Vt + (long)bh * Dc * Lc;

  const _Float16* qrA = Qh + (long)(q0 + ln) * Dc;
  const _Float16* qrB = Qh + (long)(q0 + 16 + ln) * Dc;
  const v16h qbA0 = *(const v16h*)(qrA + 16 * hi);
  const v16h qbA1 = *(const v16h*)(qrA + 32 + 16 * hi);
  const v16h qbB0 = *(const v16h*)(qrB + 16 * hi);
  const v16h qbB1 = *(const v16h*)(qrB + 32 + 16 * hi);

  v8f aA0 = {}, aA1 = {}, aA2 = {}, aA3 = {};
  v8f aB0 = {}, aB1 = {}, aB2 = {}, aB3 = {};
  float mA = -3.0e30f, lA = 0.0f, mB = -3.0e30f, lB = 0.0f;
  const int qgA = q0 + ln, qgB = q0 + 16 + ln;

  struct KF { v16h a0a, a0b, a1a, a1b; };
  auto kload = [&](KF& f, int s) {
    f.a0a = load_a_frag(Kh, Dc, s,      0,  lane);
    f.a0b = load_a_frag(Kh, Dc, s,      32, lane);
    f.a1a = load_a_frag(Kh, Dc, s + 16, 0,  lane);
    f.a1b = load_a_frag(Kh, Dc, s + 16, 32, lane);
  };

  auto step = [&](int s0, KF& cur, KF& nxt) {
    // issue next K tile + current V tile loads before any compute
    const int sn = (s0 + 32 < Lc) ? (s0 + 32) : s0;
    kload(nxt, sn);
    const long vc = s0 + 8 * hi;
    const v16h vb0 = __builtin_shufflevector(*(const v8h*)(Vh + (long)(ln)      * Lc + vc), *(const v8h*)(Vh + (long)(ln)      * Lc + vc + 16), 0,1,2,3,4,5,6,7,8,9,10,11,12,13,14,15);
    const v16h vb1 = __builtin_shufflevector(*(const v8h*)(Vh + (long)(16 + ln) * Lc + vc), *(const v8h*)(Vh + (long)(16 + ln) * Lc + vc + 16), 0,1,2,3,4,5,6,7,8,9,10,11,12,13,14,15);
    const v16h vb2 = __builtin_shufflevector(*(const v8h*)(Vh + (long)(32 + ln) * Lc + vc), *(const v8h*)(Vh + (long)(32 + ln) * Lc + vc + 16), 0,1,2,3,4,5,6,7,8,9,10,11,12,13,14,15);
    const v16h vb3 = __builtin_shufflevector(*(const v8h*)(Vh + (long)(48 + ln) * Lc + vc), *(const v8h*)(Vh + (long)(48 + ln) * Lc + vc + 16), 0,1,2,3,4,5,6,7,8,9,10,11,12,13,14,15);
    if (s0 + 64 < Lc) {   // L2 prefetch two tiles ahead (global_prefetch_b8)
      __builtin_prefetch(Kh + (long)(s0 + 64 + ln) * Dc, 0, 2);
      __builtin_prefetch(Vh + (long)(ln) * Lc + s0 + 64, 0, 2);
    }

    // S^T tiles: 8 WMMA (interleave A/B q-tiles for hazard slack)
    v8f sA0 = {}, sB0 = {}, sA1 = {}, sB1 = {};
    sA0 = WMMA16(cur.a0a, qbA0, sA0);
    sB0 = WMMA16(cur.a0a, qbB0, sB0);
    sA1 = WMMA16(cur.a1a, qbA0, sA1);
    sB1 = WMMA16(cur.a1a, qbB0, sB1);
    sA0 = WMMA16(cur.a0b, qbA1, sA0);
    sB0 = WMMA16(cur.a0b, qbB1, sB0);
    sA1 = WMMA16(cur.a1b, qbA1, sA1);
    sB1 = WMMA16(cur.a1b, qbB1, sB1);

    float vA0[8], vA1[8], vB0[8], vB1[8];
#pragma unroll
    for (int i = 0; i < 8; ++i) {
      vA0[i] = sA0[i] * SCALEc; vA1[i] = sA1[i] * SCALEc;
      vB0[i] = sB0[i] * SCALEc; vB1[i] = sB1[i] * SCALEc;
    }
    if (s0 == q0) {   // uniform branch: diagonal hits exactly one 32-key block
#pragma unroll
      for (int i = 0; i < 8; ++i) {
        const int sg0 = s0 + i + 8 * hi, sg1 = sg0 + 16;
        if (sg0 == qgA) vA0[i] = NEGc;
        if (sg1 == qgA) vA1[i] = NEGc;
        if (sg0 == qgB) vB0[i] = NEGc;
        if (sg1 == qgB) vB1[i] = NEGc;
      }
    }

    float mtA = -3.0e30f, mtB = -3.0e30f;
#pragma unroll
    for (int i = 0; i < 8; ++i) {
      mtA = fmaxf(mtA, fmaxf(vA0[i], vA1[i]));
      mtB = fmaxf(mtB, fmaxf(vB0[i], vB1[i]));
    }
    mtA = fmaxf(mtA, __shfl_xor(mtA, 16, 32));
    mtB = fmaxf(mtB, __shfl_xor(mtB, 16, 32));
    const float mnA = fmaxf(mA, mtA), mnB = fmaxf(mB, mtB);
    const float alA = __expf(mA - mnA), alB = __expf(mB - mnB);

    v16h paA, paB;
    float psA = 0.0f, psB = 0.0f;
#pragma unroll
    for (int i = 0; i < 8; ++i) {
      const float pA0 = __expf(vA0[i] - mnA), pA1 = __expf(vA1[i] - mnA);
      const float pB0 = __expf(vB0[i] - mnB), pB1 = __expf(vB1[i] - mnB);
      psA += pA0 + pA1; psB += pB0 + pB1;
      paA[i] = (_Float16)pA0; paA[8 + i] = (_Float16)pA1;
      paB[i] = (_Float16)pB0; paB[8 + i] = (_Float16)pB1;
    }
    psA += __shfl_xor(psA, 16, 32);
    psB += __shfl_xor(psB, 16, 32);
    lA = lA * alA + psA; lB = lB * alB + psB;
    mA = mnA; mB = mnB;

    aA0 *= alA; aA1 *= alA; aA2 *= alA; aA3 *= alA;
    aB0 *= alB; aB1 *= alB; aB2 *= alB; aB3 *= alB;

    // P*V : 8 WMMA, V fragments shared by both q tiles
    aA0 = WMMA16(paA, vb0, aA0);
    aB0 = WMMA16(paB, vb0, aB0);
    aA1 = WMMA16(paA, vb1, aA1);
    aB1 = WMMA16(paB, vb1, aB1);
    aA2 = WMMA16(paA, vb2, aA2);
    aB2 = WMMA16(paB, vb2, aB2);
    aA3 = WMMA16(paA, vb3, aA3);
    aB3 = WMMA16(paB, vb3, aB3);
  };

  KF f0, f1;
  kload(f0, 0);
  for (int s0 = 0; s0 < Lc; s0 += 64) {   // ping-pong: no rotation moves
    step(s0,      f0, f1);
    step(s0 + 32, f1, f0);
  }

  // ---- epilogue: normalize; acc lanes hold q = base + i + 8*hi ----
  float riA[8], riB[8];
#pragma unroll
  for (int i = 0; i < 8; ++i) {
    riA[i] = 1.0f / __shfl(lA, i + 8 * hi, 32);
    riB[i] = 1.0f / __shfl(lB, i + 8 * hi, 32);
  }

  const long b = bh >> 3, h = bh & 7;
  const long rbase = b * Lc + q0;
  v8f aa[4] = {aA0, aA1, aA2, aA3};
  v8f bb[4] = {aB0, aB1, aB2, aB3};
#pragma unroll
  for (int j = 0; j < 4; ++j)
#pragma unroll
    for (int i = 0; i < 8; ++i) {
      const long col = h * Dc + 16 * j + ln;
      O[(rbase + i + 8 * hi) * (long)DIMc + col]      = (_Float16)(aa[j][i] * riA[i]);
      O[(rbase + 16 + i + 8 * hi) * (long)DIMc + col] = (_Float16)(bb[j][i] * riB[i]);
    }
}

// ---------------- host launch ----------------
extern "C" void kernel_launch(void* const* d_in, const int* in_sizes, int n_in,
                              void* d_out, int out_size, void* d_ws, size_t ws_size,
                              hipStream_t stream) {
  const float* x  = (const float*)d_in[0];
  const float* wq = (const float*)d_in[1];
  const float* wk = (const float*)d_in[2];
  const float* wv = (const float*)d_in[3];
  const float* wo = (const float*)d_in[4];

  const int MN  = Bc * Lc;
  const int NX  = MN * DIMc;
  const int NW  = DIMc * DIMc;
  const long NQ = (long)Bc * Hc * Lc * Dc;

  _Float16* ws  = (_Float16*)d_ws;
  _Float16* xh  = ws;               ws += NX;
  _Float16* wqh = ws;               ws += NW;
  _Float16* wkh = ws;               ws += NW;
  _Float16* wvh = ws;               ws += NW;
  _Float16* woh = ws;               ws += NW;
  _Float16* Qb  = ws;               ws += NQ;
  _Float16* Kb  = ws;               ws += NQ;
  _Float16* Vt  = ws;               ws += NQ;
  _Float16* AO  = ws;               ws += NX;

  cvt_f32_f16<<<(NX + 255) / 256, 256, 0, stream>>>(x,  xh,  NX);
  cvt_f32_f16<<<(NW + 255) / 256, 256, 0, stream>>>(wq, wqh, NW);
  cvt_f32_f16<<<(NW + 255) / 256, 256, 0, stream>>>(wk, wkh, NW);
  cvt_f32_f16<<<(NW + 255) / 256, 256, 0, stream>>>(wv, wvh, NW);
  cvt_f32_f16<<<(NW + 255) / 256, 256, 0, stream>>>(wo, woh, NW);

  const int gemmBlocks = (MN / 32) * (DIMc / 32) / 4;   // 1024
  gemm_nt<0><<<gemmBlocks, 128, 0, stream>>>(xh, wqh, Qb, MN, DIMc, DIMc);
  gemm_nt<0><<<gemmBlocks, 128, 0, stream>>>(xh, wkh, Kb, MN, DIMc, DIMc);
  gemm_nt<1><<<gemmBlocks, 128, 0, stream>>>(xh, wvh, Vt, MN, DIMc, DIMc);

  // B*H*(L/32) = 2048 waves -> 512 blocks of 4 waves
  attn_diagmask<<<512, 128, 0, stream>>>(Qb, Kb, Vt, AO);

  gemm_nt<2><<<gemmBlocks, 128, 0, stream>>>(AO, woh, d_out, MN, DIMc, DIMc);
}
